// dMasifWrapper_12506944766306
// MI455X (gfx1250) — matrix operand order repaired and checked by reading, not compiled, not run on hardware
//
#include <hip/hip_runtime.h>

// ---------------------------------------------------------------------------
// dMaSIF-style forward pass for MI455X (gfx1250, wave32, WMMA).
// All N^2 pair contractions are tiled 16(M) x 32(K) and fed to
// v_wmma_f32_16x16x32_f16 (f32 accumulate). Pair weights (exp of distances)
// are built in VALU directly into the CDNA5 A-fragment layout; feature tiles
// are staged in LDS and gathered into the B-fragment layout.
// ---------------------------------------------------------------------------

typedef _Float16 v16h __attribute__((ext_vector_type(16)));
typedef float    v8f  __attribute__((ext_vector_type(8)));

constexpr int   NPTS   = 4096;
constexpr float RADIUS = 9.0f;
constexpr float REG    = 0.01f;

// Workspace layout (in floats). All offsets are multiples of 4 (16B aligned).
constexpr int NS_OFF    = 0;                       // 5 * N * 4  (smoothed normals per scale, stride 4)
constexpr int CURV_OFF  = NS_OFF + 5 * NPTS * 4;   // N * 16     (10 used)
constexpr int WSC_OFF   = CURV_OFF + NPTS * 16;    // N          (orientation scalar)
constexpr int HRAW_OFF  = WSC_OFF + NPTS;          // N * 16     (pre-GN hidden)
constexpr int STATS_OFF = HRAW_OFF + NPTS * 16;    // 16         ([0..7]=in GN, [8..15]=out GN)
constexpr int NUV_OFF   = STATS_OFF + 16;          // N * 12     (3 frame rows, stride 4)
constexpr int XI_OFF    = NUV_OFF + NPTS * 12;     // N * 16
constexpr int T_OFF     = XI_OFF + NPTS * 16;      // N * 16
constexpr int BASE_OFF  = T_OFF + NPTS * 16;       // N * 16
constexpr int HF16_OFF  = BASE_OFF + NPTS * 16;    // N * 16 halves = N*8 floats

__device__ __forceinline__ v8f wmma16(v16h a, v16h b, v8f c) {
  // D = A(16x32 f16) x B(32x16 f16) + C(16x16 f32)
  return __builtin_amdgcn_wmma_f32_16x16x32_f16(false, a, false, b, (short)0, c,
                                                false, false);
}

__device__ __forceinline__ float leaky(float v) { return v >= 0.f ? v : 0.2f * v; }
__device__ __forceinline__ float clip1(float v) { return fminf(fmaxf(v, -1.f), 1.f); }

// A-fragment k index for element t of v16h, half-group g = lane>>4.
// Layout (ISA 7.12.2, 16-bit A 16x32): g=0 holds K in {0..7, 16..23}, g=1 {8..15, 24..31}.
__device__ __forceinline__ int kmapA(int t, int g) {
  return (t & 7) + ((t >> 3) << 4) + (g << 3);
}
// B-fragment k index (dense B 32x16): same as A, g=0 -> K {0..7, 16..23}, g=1 -> {8..15, 24..31}.
__device__ __forceinline__ int kmapB(int t, int g) { return (t & 7) + ((t >> 3) << 4) + (g << 3); }

__device__ __forceinline__ void tangent_vec(float x, float y, float z,
                                            float* t1, float* t2) {
  float s = (z >= 0.f) ? 1.f : -1.f;
  float a = -1.f / (s + z);
  float b = x * y * a;
  t1[0] = 1.f + s * x * x * a; t1[1] = s * b;           t1[2] = -s * x;
  t2[0] = b;                   t2[1] = s + y * y * a;   t2[2] = -y;
}

// ---------------------------------------------------------------------------
// Curvature pass 1: u_n[s] = w_s @ normals ; ns[s] = normalize(u_n[s]).
// 16 rows per block, 4 waves split K. 5 WMMAs per k-tile (one per scale).
// ---------------------------------------------------------------------------
__global__ __launch_bounds__(128) void curv1_kernel(const float* __restrict__ verts,
                                                    const float* __restrict__ normals,
                                                    float* __restrict__ ns_out) {
  __shared__ float sV[4][32][3];
  __shared__ float sNrm[4][32][3];
  __shared__ float red[4][5][16][16];
  const int lane = threadIdx.x & 31;
  const int wave = threadIdx.x >> 5;
  const int g    = lane >> 4;
  const int r    = lane & 15;
  const int i0   = blockIdx.x * 16;
  const int i    = i0 + r;

  const float vx = verts[i * 3 + 0], vy = verts[i * 3 + 1], vz = verts[i * 3 + 2];
  const float vlen2 = vx * vx + vy * vy + vz * vz;
  const float inv2s2[5] = {0.5f, 0.125f, 1.f / 18.f, 0.02f, 0.005f};

  v8f acc[5] = {};
  for (int tt = 0; tt < 32; ++tt) {
    const int j0 = (wave * 32 + tt) * 32;
    const int j  = j0 + lane;
    sV[wave][lane][0]   = verts[j * 3 + 0];
    sV[wave][lane][1]   = verts[j * 3 + 1];
    sV[wave][lane][2]   = verts[j * 3 + 2];
    sNrm[wave][lane][0] = normals[j * 3 + 0];
    sNrm[wave][lane][1] = normals[j * 3 + 1];
    sNrm[wave][lane][2] = normals[j * 3 + 2];
    if (tt + 1 < 32) __builtin_prefetch(&verts[(j + 32) * 3], 0, 1);
    __syncthreads();

    v16h bn;
#pragma unroll
    for (int t = 0; t < 16; ++t) {
      const int k = kmapB(t, g);
      bn[t] = (r < 3) ? (_Float16)sNrm[wave][k][r] : (_Float16)0.f;
    }
    v16h af[5];
#pragma unroll
    for (int t = 0; t < 16; ++t) {
      const int k = kmapA(t, g);
      const float wx = sV[wave][k][0], wy = sV[wave][k][1], wz = sV[wave][k][2];
      float d2 = vlen2 + wx * wx + wy * wy + wz * wz -
                 2.f * (vx * wx + vy * wy + vz * wz);
      d2 = fmaxf(d2, 0.f);
#pragma unroll
      for (int s = 0; s < 5; ++s) af[s][t] = (_Float16)__expf(-d2 * inv2s2[s]);
    }
#pragma unroll
    for (int s = 0; s < 5; ++s) acc[s] = wmma16(af[s], bn, acc[s]);
    __syncthreads();
  }

#pragma unroll
  for (int s = 0; s < 5; ++s)
#pragma unroll
    for (int e = 0; e < 8; ++e) red[wave][s][e + 8 * g][r] = acc[s][e];
  __syncthreads();

  if (threadIdx.x < 80) {
    const int s = threadIdx.x >> 4;
    const int m = threadIdx.x & 15;
    float u0 = 0.f, u1 = 0.f, u2 = 0.f;
    for (int w = 0; w < 4; ++w) {
      u0 += red[w][s][m][0]; u1 += red[w][s][m][1]; u2 += red[w][s][m][2];
    }
    const float inv = 1.f / fmaxf(sqrtf(u0 * u0 + u1 * u1 + u2 * u2), 1e-12f);
    const int gi = i0 + m;
    ns_out[(s * NPTS + gi) * 4 + 0] = u0 * inv;
    ns_out[(s * NPTS + gi) * 4 + 1] = u1 * inv;
    ns_out[(s * NPTS + gi) * 4 + 2] = u2 * inv;
  }
}

// Feature column for curvature pass 2 (B matrix, 25 used columns):
// 0-2: v ; 3-5: ns ; 6-14: v_a*v_b ; 15-23: v_a*ns_b ; 24: 1 (gives s0 for free).
__device__ __forceinline__ float colval(int n, const float* v, const float* nsv) {
  if (n < 3) return v[n];
  if (n < 6) return nsv[n - 3];
  if (n < 15) { int q = n - 6;  return v[q / 3] * v[q % 3]; }
  if (n < 24) { int q = n - 15; return v[q / 3] * nsv[q % 3]; }
  return (n == 24) ? 1.f : 0.f;
}

// ---------------------------------------------------------------------------
// Curvature pass 2: [sx sn sxx sxn s0] = w_s @ F(25 cols) per scale, then the
// per-point tangent-frame 2x2 solve in the epilogue. 10 WMMAs per k-tile.
// ---------------------------------------------------------------------------
__global__ __launch_bounds__(128) void curv2_kernel(const float* __restrict__ verts,
                                                    const float* __restrict__ ns_in,
                                                    float* __restrict__ curv) {
  __shared__ float sV[4][32][3];
  __shared__ float sNs[4][5][32][3];
  __shared__ float red[4][5][16][32];
  const int lane = threadIdx.x & 31;
  const int wave = threadIdx.x >> 5;
  const int g    = lane >> 4;
  const int r    = lane & 15;
  const int i0   = blockIdx.x * 16;
  const int i    = i0 + r;

  const float vx = verts[i * 3 + 0], vy = verts[i * 3 + 1], vz = verts[i * 3 + 2];
  const float vlen2 = vx * vx + vy * vy + vz * vz;
  const float inv2s2[5] = {0.5f, 0.125f, 1.f / 18.f, 0.02f, 0.005f};

  v8f acc0[5] = {}, acc1[5] = {};
  for (int tt = 0; tt < 32; ++tt) {
    const int j0 = (wave * 32 + tt) * 32;
    const int j  = j0 + lane;
    sV[wave][lane][0] = verts[j * 3 + 0];
    sV[wave][lane][1] = verts[j * 3 + 1];
    sV[wave][lane][2] = verts[j * 3 + 2];
#pragma unroll
    for (int s = 0; s < 5; ++s) {
      sNs[wave][s][lane][0] = ns_in[(s * NPTS + j) * 4 + 0];
      sNs[wave][s][lane][1] = ns_in[(s * NPTS + j) * 4 + 1];
      sNs[wave][s][lane][2] = ns_in[(s * NPTS + j) * 4 + 2];
    }
    __syncthreads();

    float d2v[16];
#pragma unroll
    for (int t = 0; t < 16; ++t) {
      const int k = kmapA(t, g);
      const float wx = sV[wave][k][0], wy = sV[wave][k][1], wz = sV[wave][k][2];
      float d2 = vlen2 + wx * wx + wy * wy + wz * wz -
                 2.f * (vx * wx + vy * wy + vz * wz);
      d2v[t] = fmaxf(d2, 0.f);
    }
#pragma unroll
    for (int s = 0; s < 5; ++s) {
      v16h af, b0, b1;
#pragma unroll
      for (int t = 0; t < 16; ++t) af[t] = (_Float16)__expf(-d2v[t] * inv2s2[s]);
#pragma unroll
      for (int t = 0; t < 16; ++t) {
        const int k = kmapB(t, g);
        b0[t] = (_Float16)colval(r,      sV[wave][k], sNs[wave][s][k]);
        b1[t] = (_Float16)colval(r + 16, sV[wave][k], sNs[wave][s][k]);
      }
      acc0[s] = wmma16(af, b0, acc0[s]);
      acc1[s] = wmma16(af, b1, acc1[s]);
    }
    __syncthreads();
  }

#pragma unroll
  for (int s = 0; s < 5; ++s)
#pragma unroll
    for (int e = 0; e < 8; ++e) {
      red[wave][s][e + 8 * g][r]      = acc0[s][e];
      red[wave][s][e + 8 * g][r + 16] = acc1[s][e];
    }
  __syncthreads();

  if (threadIdx.x < 80) {
    const int s  = threadIdx.x >> 4;
    const int m  = threadIdx.x & 15;
    const int gi = i0 + m;
    float col[25];
#pragma unroll
    for (int n = 0; n < 25; ++n) {
      float a = 0.f;
      for (int w = 0; w < 4; ++w) a += red[w][s][m][n];
      col[n] = a;
    }
    const float v[3]   = {verts[gi * 3 + 0], verts[gi * 3 + 1], verts[gi * 3 + 2]};
    const float nsv[3] = {ns_in[(s * NPTS + gi) * 4 + 0],
                          ns_in[(s * NPTS + gi) * 4 + 1],
                          ns_in[(s * NPTS + gi) * 4 + 2]};
    const float* sx = &col[0];
    const float* sn = &col[3];
    const float  s0 = col[24];
    float t1[3], t2[3];
    tangent_vec(nsv[0], nsv[1], nsv[2], t1, t2);
    float cxx[9], cxn[9];
#pragma unroll
    for (int a = 0; a < 3; ++a)
#pragma unroll
      for (int b = 0; b < 3; ++b) {
        cxx[a * 3 + b] = col[6 + a * 3 + b]  - v[a] * sx[b] - sx[a] * v[b]   + s0 * v[a] * v[b];
        cxn[a * 3 + b] = col[15 + a * 3 + b] - v[a] * sn[b] - sx[a] * nsv[b] + s0 * v[a] * nsv[b];
      }
    auto quad = [](const float* u, const float* M, const float* w2) {
      float a = 0.f;
#pragma unroll
      for (int p = 0; p < 3; ++p)
#pragma unroll
        for (int q = 0; q < 3; ++q) a += u[p] * M[p * 3 + q] * w2[q];
      return a;
    };
    const float ppt00 = quad(t1, cxx, t1) + REG;
    const float ppt01 = quad(t1, cxx, t2);
    const float ppt10 = quad(t2, cxx, t1);
    const float ppt11 = quad(t2, cxx, t2) + REG;
    const float pqt00 = quad(t1, cxn, t1);
    const float pqt01 = quad(t1, cxn, t2);
    const float pqt10 = quad(t2, cxn, t1);
    const float pqt11 = quad(t2, cxn, t2);
    const float id = 1.f / (ppt00 * ppt11 - ppt01 * ppt10);
    const float S00 = (ppt11 * pqt00 - ppt01 * pqt10) * id;
    const float S01 = (ppt11 * pqt01 - ppt01 * pqt11) * id;
    const float S10 = (ppt00 * pqt10 - ppt10 * pqt00) * id;
    const float S11 = (ppt00 * pqt11 - ppt10 * pqt01) * id;
    curv[gi * 16 + 2 * s]     = clip1(S00 + S11);
    curv[gi * 16 + 2 * s + 1] = clip1(S00 * S11 - S01 * S10);
  }
}

// ---------------------------------------------------------------------------
// Per-point MLPs: orientation score, net_in hidden (pre-GN), linear transform.
// ---------------------------------------------------------------------------
__global__ __launch_bounds__(128) void mlp1_kernel(
    const float* __restrict__ x, const float* __restrict__ curv,
    const float* __restrict__ os_w1, const float* __restrict__ os_b1,
    const float* __restrict__ os_w2, const float* __restrict__ os_b2,
    const float* __restrict__ ni_w1, const float* __restrict__ ni_b1,
    const float* __restrict__ ni_w2, const float* __restrict__ ni_b2,
    const float* __restrict__ lt_w, const float* __restrict__ lt_b,
    float* __restrict__ wsc, float* __restrict__ hraw, float* __restrict__ base) {
  const int i = blockIdx.x * blockDim.x + threadIdx.x;
  if (i >= NPTS) return;
  float xf[26];
#pragma unroll
  for (int c = 0; c < 16; ++c) xf[c] = x[i * 16 + c];
#pragma unroll
  for (int c = 0; c < 10; ++c) xf[16 + c] = curv[i * 16 + c];

  float wsum = os_b2[0];
  for (int o = 0; o < 16; ++o) {
    float a = os_b1[o];
    for (int c = 0; c < 26; ++c) a += xf[c] * os_w1[o * 26 + c];
    wsum += leaky(a) * os_w2[o];
  }
  wsc[i] = wsum;

  float h1[16];
  for (int o = 0; o < 16; ++o) {
    float a = ni_b1[o];
    for (int c = 0; c < 26; ++c) a += xf[c] * ni_w1[o * 26 + c];
    h1[o] = leaky(a);
  }
  for (int o = 0; o < 16; ++o) {
    float a = ni_b2[o];
    for (int c = 0; c < 16; ++c) a += h1[c] * ni_w2[o * 16 + c];
    hraw[i * 16 + o] = leaky(a);
  }
  for (int o = 0; o < 16; ++o) {
    float a = lt_b[o];
    for (int c = 0; c < 26; ++c) a += xf[c] * lt_w[o * 26 + c];
    base[i * 16 + o] = a;
  }
}

// Deterministic group-norm statistics (4 groups x 4 channels over N points).
__global__ __launch_bounds__(256) void gn_stats_kernel(const float* __restrict__ src,
                                                       float* __restrict__ stats) {
  __shared__ float rs[4][256];
  __shared__ float rq[4][256];
  float s[4] = {0, 0, 0, 0}, q[4] = {0, 0, 0, 0};
  for (int i = threadIdx.x; i < NPTS; i += 256) {
#pragma unroll
    for (int gg = 0; gg < 4; ++gg)
#pragma unroll
      for (int c = 0; c < 4; ++c) {
        const float v = src[i * 16 + gg * 4 + c];
        s[gg] += v; q[gg] += v * v;
      }
  }
#pragma unroll
  for (int gg = 0; gg < 4; ++gg) { rs[gg][threadIdx.x] = s[gg]; rq[gg][threadIdx.x] = q[gg]; }
  __syncthreads();
  for (int off = 128; off > 0; off >>= 1) {
    if (threadIdx.x < off)
#pragma unroll
      for (int gg = 0; gg < 4; ++gg) {
        rs[gg][threadIdx.x] += rs[gg][threadIdx.x + off];
        rq[gg][threadIdx.x] += rq[gg][threadIdx.x + off];
      }
    __syncthreads();
  }
  if (threadIdx.x < 4) {
    const float inv = 1.f / (4.f * (float)NPTS);
    const float mean = rs[threadIdx.x][0] * inv;
    stats[threadIdx.x]     = mean;
    stats[4 + threadIdx.x] = rq[threadIdx.x][0] * inv - mean * mean;
  }
}

// Apply input group-norm and emit f16 copy of h (B matrix for the conv stage).
__global__ __launch_bounds__(128) void gn_in_kernel(const float* __restrict__ hraw,
                                                    const float* __restrict__ stats,
                                                    const float* __restrict__ gw,
                                                    const float* __restrict__ gb,
                                                    _Float16* __restrict__ hf) {
  const int i = blockIdx.x * blockDim.x + threadIdx.x;
  if (i >= NPTS) return;
#pragma unroll
  for (int c = 0; c < 16; ++c) {
    const int gg = c >> 2;
    const float v = (hraw[i * 16 + c] - stats[gg]) * rsqrtf(stats[4 + gg] + 1e-5f);
    hf[i * 16 + c] = (_Float16)(v * gw[c] + gb[c]);
  }
}

// ---------------------------------------------------------------------------
// Orientation: ov3 = W @ (w*p) - (W @ w) * p ; build local frames nuv.
// One WMMA per k-tile (B cols: w*p x3, w).
// ---------------------------------------------------------------------------
__global__ __launch_bounds__(128) void orient_kernel(const float* __restrict__ verts,
                                                     const float* __restrict__ vnormals,
                                                     const float* __restrict__ wsc,
                                                     float* __restrict__ nuv) {
  __shared__ float sP[4][32][3];
  __shared__ float sN[4][32][3];
  __shared__ float sW[4][32];
  __shared__ float red[4][16][16];
  const int lane = threadIdx.x & 31;
  const int wave = threadIdx.x >> 5;
  const int g    = lane >> 4;
  const int r    = lane & 15;
  const int i0   = blockIdx.x * 16;
  const int i    = i0 + r;
  const float invR = 1.f / RADIUS;

  const float pix = verts[i * 3 + 0] * invR, piy = verts[i * 3 + 1] * invR,
              piz = verts[i * 3 + 2] * invR;
  const float nix = vnormals[i * 3 + 0], niy = vnormals[i * 3 + 1],
              niz = vnormals[i * 3 + 2];

  v8f acc = {};
  for (int tt = 0; tt < 32; ++tt) {
    const int j0 = (wave * 32 + tt) * 32;
    const int j  = j0 + lane;
    sP[wave][lane][0] = verts[j * 3 + 0] * invR;
    sP[wave][lane][1] = verts[j * 3 + 1] * invR;
    sP[wave][lane][2] = verts[j * 3 + 2] * invR;
    sN[wave][lane][0] = vnormals[j * 3 + 0];
    sN[wave][lane][1] = vnormals[j * 3 + 1];
    sN[wave][lane][2] = vnormals[j * 3 + 2];
    sW[wave][lane]    = wsc[j];
    __syncthreads();

    v16h bf;
#pragma unroll
    for (int t = 0; t < 16; ++t) {
      const int k = kmapB(t, g);
      const float val = (r < 3) ? sW[wave][k] * sP[wave][k][r]
                                : ((r == 3) ? sW[wave][k] : 0.f);
      bf[t] = (_Float16)val;
    }
    v16h af;
#pragma unroll
    for (int t = 0; t < 16; ++t) {
      const int k = kmapA(t, g);
      const float dx = sP[wave][k][0] - pix, dy = sP[wave][k][1] - piy,
                  dz = sP[wave][k][2] - piz;
      const float d2 = dx * dx + dy * dy + dz * dz;
      const float nd = nix * sN[wave][k][0] + niy * sN[wave][k][1] + niz * sN[wave][k][2];
      const float f  = 2.f - nd;
      af[t] = (_Float16)__expf(-d2 * f * f);
    }
    acc = wmma16(af, bf, acc);
    __syncthreads();
  }

#pragma unroll
  for (int e = 0; e < 8; ++e) red[wave][e + 8 * g][r] = acc[e];
  __syncthreads();

  if (threadIdx.x < 16) {
    const int m  = threadIdx.x;
    const int gi = i0 + m;
    float o0 = 0.f, o1 = 0.f, o2 = 0.f, sw = 0.f;
    for (int w = 0; w < 4; ++w) {
      o0 += red[w][m][0]; o1 += red[w][m][1]; o2 += red[w][m][2]; sw += red[w][m][3];
    }
    const float px = verts[gi * 3 + 0] * invR, py = verts[gi * 3 + 1] * invR,
                pz = verts[gi * 3 + 2] * invR;
    o0 -= sw * px; o1 -= sw * py; o2 -= sw * pz;
    const float nx = vnormals[gi * 3 + 0], ny = vnormals[gi * 3 + 1],
                nz = vnormals[gi * 3 + 2];
    float t1[3], t2[3];
    tangent_vec(nx, ny, nz, t1, t2);
    const float ov0 = t1[0] * o0 + t1[1] * o1 + t1[2] * o2 + 1e-5f;
    const float ov1 = t2[0] * o0 + t2[1] * o1 + t2[2] * o2 + 1e-5f;
    const float inv = rsqrtf(ov0 * ov0 + ov1 * ov1);
    const float ex = ov0 * inv, ey = ov1 * inv;
    float* dst = nuv + gi * 12;
    dst[0] = nx; dst[1] = ny; dst[2] = nz; dst[3] = 0.f;
    dst[4] = ex * t1[0] + ey * t2[0];
    dst[5] = ex * t1[1] + ey * t2[1];
    dst[6] = ex * t1[2] + ey * t2[2];
    dst[7] = 0.f;
    dst[8]  = -ey * t1[0] + ex * t2[0];
    dst[9]  = -ey * t1[1] + ex * t2[1];
    dst[10] = -ey * t1[2] + ex * t2[2];
    dst[11] = 0.f;
  }
}

// ---------------------------------------------------------------------------
// Tangent conv: xi[i,h] = sum_c a2[h,c]*G_c(i,h) + b2[h]*G_8(i,h), where
// G_c = (win*relu_c) @ h  and  G_8 = win @ h. 9 WMMAs per k-tile, shared B.
// ---------------------------------------------------------------------------
__global__ __launch_bounds__(128) void conv_kernel(
    const float* __restrict__ verts, const float* __restrict__ vnormals,
    const float* __restrict__ nuv, const _Float16* __restrict__ hf,
    const float* __restrict__ cv_a1, const float* __restrict__ cv_b1,
    const float* __restrict__ cv_a2, const float* __restrict__ cv_b2,
    float* __restrict__ xi) {
  __shared__ float sP[4][32][3];
  __shared__ float sN[4][32][3];
  __shared__ __align__(16) _Float16 sH[4][32][16];
  __shared__ float red[4][9][16][16];
  const int lane = threadIdx.x & 31;
  const int wave = threadIdx.x >> 5;
  const int g    = lane >> 4;
  const int r    = lane & 15;
  const int i0   = blockIdx.x * 16;
  const int i    = i0 + r;
  const float invR = 1.f / RADIUS;

  const float pi0 = verts[i * 3 + 0] * invR, pi1 = verts[i * 3 + 1] * invR,
              pi2 = verts[i * 3 + 2] * invR;
  float nv[9];
#pragma unroll
  for (int c = 0; c < 9; ++c) nv[c] = nuv[i * 12 + (c / 3) * 4 + (c % 3)];
  float a1[8][3], b1v[8];
#pragma unroll
  for (int c = 0; c < 8; ++c) {
    a1[c][0] = cv_a1[c * 3 + 0];
    a1[c][1] = cv_a1[c * 3 + 1];
    a1[c][2] = cv_a1[c * 3 + 2];
    b1v[c]   = cv_b1[c];
  }

  v8f acc[9] = {};
  for (int tt = 0; tt < 32; ++tt) {
    const int j0 = (wave * 32 + tt) * 32;
    const int j  = j0 + lane;
    sP[wave][lane][0] = verts[j * 3 + 0] * invR;
    sP[wave][lane][1] = verts[j * 3 + 1] * invR;
    sP[wave][lane][2] = verts[j * 3 + 2] * invR;
    sN[wave][lane][0] = vnormals[j * 3 + 0];
    sN[wave][lane][1] = vnormals[j * 3 + 1];
    sN[wave][lane][2] = vnormals[j * 3 + 2];
    *(uint4*)&sH[wave][lane][0] = *(const uint4*)&hf[j * 16 + 0];
    *(uint4*)&sH[wave][lane][8] = *(const uint4*)&hf[j * 16 + 8];
    __syncthreads();

    v16h bh;
#pragma unroll
    for (int t = 0; t < 16; ++t) bh[t] = sH[wave][kmapB(t, g)][r];

    v16h afr[9];
#pragma unroll
    for (int t = 0; t < 16; ++t) {
      const int k = kmapA(t, g);
      const float dx = sP[wave][k][0] - pi0, dy = sP[wave][k][1] - pi1,
                  dz = sP[wave][k][2] - pi2;
      const float d2 = dx * dx + dy * dy + dz * dz;
      const float nd = nv[0] * sN[wave][k][0] + nv[1] * sN[wave][k][1] +
                       nv[2] * sN[wave][k][2];
      const float f2  = 2.f - nd;
      const float win = __expf(-d2 * f2 * f2);
      const float X0 = nv[0] * dx + nv[1] * dy + nv[2] * dz;
      const float X1 = nv[3] * dx + nv[4] * dy + nv[5] * dz;
      const float X2 = nv[6] * dx + nv[7] * dy + nv[8] * dz;
#pragma unroll
      for (int c = 0; c < 8; ++c) {
        const float rc = fmaxf(a1[c][0] * X0 + a1[c][1] * X1 + a1[c][2] * X2 + b1v[c], 0.f);
        afr[c][t] = (_Float16)(win * rc);
      }
      afr[8][t] = (_Float16)win;
    }
#pragma unroll
    for (int c = 0; c < 9; ++c) acc[c] = wmma16(afr[c], bh, acc[c]);
    __syncthreads();
  }

#pragma unroll
  for (int c = 0; c < 9; ++c)
#pragma unroll
    for (int e = 0; e < 8; ++e) red[wave][c][e + 8 * g][r] = acc[c][e];
  __syncthreads();

  for (int pair = threadIdx.x; pair < 256; pair += 128) {
    const int m  = pair >> 4;
    const int hh = pair & 15;
    float G[9];
#pragma unroll
    for (int c = 0; c < 9; ++c) {
      float a = 0.f;
      for (int w = 0; w < 4; ++w) a += red[w][c][m][hh];
      G[c] = a;
    }
    float val = cv_b2[hh] * G[8];
#pragma unroll
    for (int c = 0; c < 8; ++c) val += cv_a2[hh * 8 + c] * G[c];
    xi[(i0 + m) * 16 + hh] = val;
  }
}

// net_out MLP (two leaky 16x16 layers), pre-GN.
__global__ __launch_bounds__(128) void mlp2_kernel(const float* __restrict__ xi,
                                                   const float* __restrict__ no_w1,
                                                   const float* __restrict__ no_b1,
                                                   const float* __restrict__ no_w2,
                                                   const float* __restrict__ no_b2,
                                                   float* __restrict__ tout) {
  const int i = blockIdx.x * blockDim.x + threadIdx.x;
  if (i >= NPTS) return;
  float h1[16];
  for (int o = 0; o < 16; ++o) {
    float a = no_b1[o];
    for (int c = 0; c < 16; ++c) a += xi[i * 16 + c] * no_w1[o * 16 + c];
    h1[o] = leaky(a);
  }
  for (int o = 0; o < 16; ++o) {
    float a = no_b2[o];
    for (int c = 0; c < 16; ++c) a += h1[c] * no_w2[o * 16 + c];
    tout[i * 16 + o] = leaky(a);
  }
}

// Output GN + last-layers MLP + residual linear term.
__global__ __launch_bounds__(128) void final_kernel(
    const float* __restrict__ t, const float* __restrict__ stats2,
    const float* __restrict__ gw, const float* __restrict__ gb,
    const float* __restrict__ ll_w1, const float* __restrict__ ll_b1,
    const float* __restrict__ ll_w2, const float* __restrict__ ll_b2,
    const float* __restrict__ base, float* __restrict__ out) {
  const int i = blockIdx.x * blockDim.x + threadIdx.x;
  if (i >= NPTS) return;
  float gnv[16];
#pragma unroll
  for (int c = 0; c < 16; ++c) {
    const int gg = c >> 2;
    gnv[c] = (t[i * 16 + c] - stats2[gg]) * rsqrtf(stats2[4 + gg] + 1e-5f) * gw[c] + gb[c];
  }
  float q[16];
  for (int o = 0; o < 16; ++o) {
    float a = ll_b1[o];
    for (int c = 0; c < 16; ++c) a += gnv[c] * ll_w1[o * 16 + c];
    q[o] = fmaxf(a, 0.f);
  }
  for (int o = 0; o < 16; ++o) {
    float a = ll_b2[o];
    for (int c = 0; c < 16; ++c) a += q[c] * ll_w2[o * 16 + c];
    out[i * 16 + o] = base[i * 16 + o] + a;
  }
}

extern "C" void kernel_launch(void* const* d_in, const int* in_sizes, int n_in,
                              void* d_out, int out_size, void* d_ws, size_t ws_size,
                              hipStream_t stream) {
  (void)in_sizes; (void)n_in; (void)out_size; (void)ws_size;
  const float* verts    = (const float*)d_in[0];
  const float* vnormals = (const float*)d_in[1];
  const float* x        = (const float*)d_in[2];
  const float* os_w1    = (const float*)d_in[3];
  const float* os_b1    = (const float*)d_in[4];
  const float* os_w2    = (const float*)d_in[5];
  const float* os_b2    = (const float*)d_in[6];
  const float* ni_w1    = (const float*)d_in[7];
  const float* ni_b1    = (const float*)d_in[8];
  const float* ni_w2    = (const float*)d_in[9];
  const float* ni_b2    = (const float*)d_in[10];
  const float* gn_in_w  = (const float*)d_in[11];
  const float* gn_in_b  = (const float*)d_in[12];
  const float* cv_a1    = (const float*)d_in[13];
  const float* cv_b1    = (const float*)d_in[14];
  const float* cv_a2    = (const float*)d_in[15];
  const float* cv_b2    = (const float*)d_in[16];
  const float* no_w1    = (const float*)d_in[17];
  const float* no_b1    = (const float*)d_in[18];
  const float* no_w2    = (const float*)d_in[19];
  const float* no_b2    = (const float*)d_in[20];
  const float* gn_out_w = (const float*)d_in[21];
  const float* gn_out_b = (const float*)d_in[22];
  const float* ll_w1    = (const float*)d_in[23];
  const float* ll_b1    = (const float*)d_in[24];
  const float* ll_w2    = (const float*)d_in[25];
  const float* ll_b2    = (const float*)d_in[26];
  const float* lt_w     = (const float*)d_in[27];
  const float* lt_b     = (const float*)d_in[28];

  float* ws    = (float*)d_ws;
  float* ns    = ws + NS_OFF;
  float* curv  = ws + CURV_OFF;
  float* wsc   = ws + WSC_OFF;
  float* hraw  = ws + HRAW_OFF;
  float* stats = ws + STATS_OFF;
  float* nuvb  = ws + NUV_OFF;
  float* xib   = ws + XI_OFF;
  float* tb    = ws + T_OFF;
  float* baseb = ws + BASE_OFF;
  _Float16* hf = (_Float16*)(ws + HF16_OFF);

  const dim3 b128(128), b256(256);
  curv1_kernel<<<NPTS / 16, b128, 0, stream>>>(verts, vnormals, ns);
  curv2_kernel<<<NPTS / 16, b128, 0, stream>>>(verts, ns, curv);
  mlp1_kernel<<<NPTS / 128, b128, 0, stream>>>(x, curv, os_w1, os_b1, os_w2, os_b2,
                                               ni_w1, ni_b1, ni_w2, ni_b2, lt_w, lt_b,
                                               wsc, hraw, baseb);
  gn_stats_kernel<<<1, b256, 0, stream>>>(hraw, stats);
  gn_in_kernel<<<NPTS / 128, b128, 0, stream>>>(hraw, stats, gn_in_w, gn_in_b, hf);
  orient_kernel<<<NPTS / 16, b128, 0, stream>>>(verts, vnormals, wsc, nuvb);
  conv_kernel<<<NPTS / 16, b128, 0, stream>>>(verts, vnormals, nuvb, hf,
                                              cv_a1, cv_b1, cv_a2, cv_b2, xib);
  mlp2_kernel<<<NPTS / 128, b128, 0, stream>>>(xib, no_w1, no_b1, no_w2, no_b2, tb);
  gn_stats_kernel<<<1, b256, 0, stream>>>(tb, stats + 8);
  final_kernel<<<NPTS / 128, b128, 0, stream>>>(tb, stats + 8, gn_out_w, gn_out_b,
                                                ll_w1, ll_b1, ll_w2, ll_b2, baseb,
                                                (float*)d_out);
}
